// Token_QK_Attention_86260123173256
// MI455X (gfx1250) — matrix operand order repaired and checked by reading, not compiled, not run on hardware
//
#include <hip/hip_runtime.h>
#include <hip/hip_bf16.h>

typedef _Float16 v8h  __attribute__((ext_vector_type(8)));
typedef _Float16 v16h __attribute__((ext_vector_type(16)));
typedef float    v8f  __attribute__((ext_vector_type(8)));

#define T_      4
#define B_      16
#define C_      256
#define N_      1024
#define HEADS_  8
#define DHEAD_  32
#define BCN_    (B_ * C_ * N_)      /* 4,194,304 */
#define TBCN_   ((size_t)T_ * BCN_) /* 16,777,216 */
#define BN_EPS_ 1e-5f

// -------------------------------------------------------------------------
// Prep: per-channel BN fold (and proj bias fold): out = acc*inv[c] + shift[c]
// -------------------------------------------------------------------------
__global__ void snn_prep_channels_k(
    const float* __restrict__ qg, const float* __restrict__ qb,
    const float* __restrict__ qm, const float* __restrict__ qv,
    const float* __restrict__ kg, const float* __restrict__ kb,
    const float* __restrict__ km, const float* __restrict__ kv,
    const float* __restrict__ pg, const float* __restrict__ pb,
    const float* __restrict__ pm, const float* __restrict__ pv,
    const float* __restrict__ proj_b, float* __restrict__ params) {
  int c = threadIdx.x;
  if (c < C_) {
    float qi = qg[c] * rsqrtf(qv[c] + BN_EPS_);
    params[0 * C_ + c] = qi;
    params[1 * C_ + c] = qb[c] - qm[c] * qi;
    float ki = kg[c] * rsqrtf(kv[c] + BN_EPS_);
    params[2 * C_ + c] = ki;
    params[3 * C_ + c] = kb[c] - km[c] * ki;
    float pi = pg[c] * rsqrtf(pv[c] + BN_EPS_);
    params[4 * C_ + c] = pi;
    // BN(conv + b) = conv*pi + (beta - mean*pi + b*pi)
    params[5 * C_ + c] = pb[c] - pm[c] * pi + proj_b[c] * pi;
  }
}

// -------------------------------------------------------------------------
// Prep: fp32 -> fp16 weight conversion (q_w, k_w, proj_w)
// -------------------------------------------------------------------------
__global__ void snn_convert_w_k(const float* __restrict__ wq,
                                const float* __restrict__ wk,
                                const float* __restrict__ wp,
                                _Float16* __restrict__ q16,
                                _Float16* __restrict__ k16,
                                _Float16* __restrict__ p16) {
  int i = blockIdx.x * blockDim.x + threadIdx.x;  // 0..C*C-1
  const float* s = (blockIdx.y == 0) ? wq : (blockIdx.y == 1) ? wk : wp;
  _Float16*    d = (blockIdx.y == 0) ? q16 : (blockIdx.y == 1) ? k16 : p16;
  d[i] = (_Float16)s[i];
}

// -------------------------------------------------------------------------
// Input LIF: mem = 0.5*mem + x_t; spike = (mem >= 1); hard reset. -> fp16 {0,1}
// -------------------------------------------------------------------------
__global__ void snn_lif_input_k(const float* __restrict__ x,
                                _Float16* __restrict__ xs) {
  int i = blockIdx.x * blockDim.x + threadIdx.x;  // over B*C*N
  float mem = 0.f;
#pragma unroll
  for (int t = 0; t < T_; ++t) {
    float v = x[(size_t)t * BCN_ + i];
    mem = mem * 0.5f + v;
    float s = (mem >= 1.0f) ? 1.0f : 0.0f;
    xs[(size_t)t * BCN_ + i] = (_Float16)s;
    mem *= (1.0f - s);
  }
}

// -------------------------------------------------------------------------
// Batched GEMM + per-channel scale/shift epilogue.
// D[slice](128x128 block tile) = W(256x256) x X[slice](256x1024)
// 8 waves: each wave owns 32 rows x 64 cols = 2x4 WMMA 16x16 tiles.
// -------------------------------------------------------------------------
template <bool OUT_F32>
__global__ __launch_bounds__(256) void snn_gemm_bn_k(
    const _Float16* __restrict__ Wm,  // C x C, row-major fp16
    const _Float16* __restrict__ X,   // (T*B) slices of C x N fp16
    const float* __restrict__ inv, const float* __restrict__ shift,
    void* __restrict__ outv) {
  constexpr int LDA = 72;  // 64 + 8 pad (halves)
  constexpr int LDB = 72;
  __shared__ _Float16 lds_a[128 * LDA];  // [row][k]     (k contiguous)
  __shared__ _Float16 lds_b[128 * LDB];  // [col][k]     (transposed X)

  const int tid = threadIdx.x;
  const int wave = tid >> 5, lane = tid & 31;
  const int wr = wave & 3;   // 32-row strip within 128
  const int wc = wave >> 2;  // 64-col strip within 128
  const int lane15 = lane & 15;
  const int hi = lane >> 4;  // lane half selects K sub-range / M offset

  const int col0 = blockIdx.x * 128;
  const int row0 = blockIdx.y * 128;
  const size_t slice = blockIdx.z;
  const _Float16* Xs = X + slice * (size_t)(C_ * N_);

  v8f acc[2][4];
#pragma unroll
  for (int a = 0; a < 2; ++a)
#pragma unroll
    for (int b = 0; b < 4; ++b) acc[a][b] = (v8f){0, 0, 0, 0, 0, 0, 0, 0};

  for (int kk = 0; kk < C_; kk += 64) {
    // --- A chunk: 128 rows x 64 K, row-major, vec8 loads ---
#pragma unroll
    for (int j = 0; j < 4; ++j) {
      int lin = j * 2048 + tid * 8;  // halves
      int r = lin >> 6, k = lin & 63;
      *(v8h*)(lds_a + r * LDA + k) =
          *(const v8h*)(Wm + (size_t)(row0 + r) * C_ + kk + k);
    }
    // --- X chunk: 64 K x 128 cols, transposed into [col][k] ---
#pragma unroll
    for (int j = 0; j < 4; ++j) {
      int lin = j * 2048 + tid * 8;
      int r = lin >> 7, n = lin & 127;
      v8h g = *(const v8h*)(Xs + (size_t)(kk + r) * N_ + col0 + n);
#pragma unroll
      for (int e = 0; e < 8; ++e) lds_b[(n + e) * LDB + r] = g[e];
    }
    // Prefetch next X chunk into cache hierarchy.
    if (kk + 64 < C_) {
      __builtin_prefetch(Xs + (size_t)(kk + 64 + (tid >> 2)) * N_ + col0 +
                             (tid & 3) * 32, 0, 1);
    }
    __syncthreads();

#pragma unroll
    for (int ks = 0; ks < 64; ks += 32) {
      v16h afrag[2], bfrag[4];
      const int kloA = hi ? 8 : 0;    // A: lanes 16-31 hold K 8..15 / 24..31
      const int kloB = hi ? 8 : 0;    // B: lanes 16-31 hold K 8..15 / 24..31
#pragma unroll
      for (int tr = 0; tr < 2; ++tr) {
        const _Float16* pa =
            lds_a + (wr * 32 + tr * 16 + lane15) * LDA + ks + kloA;
        v8h lo = *(const v8h*)pa;         // K = klo .. klo+7
        v8h hh = *(const v8h*)(pa + 16);  // K = klo+16 .. klo+23
        afrag[tr] = __builtin_shufflevector(lo, hh, 0, 1, 2, 3, 4, 5, 6, 7, 8,
                                            9, 10, 11, 12, 13, 14, 15);
      }
#pragma unroll
      for (int tc = 0; tc < 4; ++tc) {
        const _Float16* pb =
            lds_b + (wc * 64 + tc * 16 + lane15) * LDB + ks + kloB;
        v8h lo = *(const v8h*)pb;        // K = klo .. klo+7
        v8h hh = *(const v8h*)(pb + 16);  // K = klo+16 .. klo+23
        bfrag[tc] = __builtin_shufflevector(lo, hh, 0, 1, 2, 3, 4, 5, 6, 7, 8,
                                            9, 10, 11, 12, 13, 14, 15);
      }
#pragma unroll
      for (int tr = 0; tr < 2; ++tr)
#pragma unroll
        for (int tc = 0; tc < 4; ++tc)
          acc[tr][tc] = __builtin_amdgcn_wmma_f32_16x16x32_f16(
              false, afrag[tr], false, bfrag[tc], (short)0, acc[tr][tc], false,
              false);
    }
    __syncthreads();
  }

  // Epilogue: per-channel scale/shift, C/D layout: lane=N col, vgpr r -> M row.
#pragma unroll
  for (int tr = 0; tr < 2; ++tr) {
#pragma unroll
    for (int r = 0; r < 8; ++r) {
      int cout = row0 + wr * 32 + tr * 16 + r + hi * 8;
      float iv = inv[cout], sh = shift[cout];
#pragma unroll
      for (int tc = 0; tc < 4; ++tc) {
        int n = col0 + wc * 64 + tc * 16 + lane15;
        float v = acc[tr][tc][r] * iv + sh;
        size_t oidx = slice * (size_t)(C_ * N_) + (size_t)cout * N_ + n;
        if (OUT_F32)
          ((float*)outv)[oidx] = v;
        else
          ((_Float16*)outv)[oidx] = (_Float16)v;
      }
    }
  }
}

// -------------------------------------------------------------------------
// Fused: q-LIF -> Wei temporal mix -> per-head d-reduction -> attn-LIF ->
//        k-LIF -> y = attn * k   (one thread per (b, head, n))
// -------------------------------------------------------------------------
__global__ void snn_attn_k(const _Float16* __restrict__ qpre,
                           const _Float16* __restrict__ kpre,
                           const float* __restrict__ alpha_p,
                           _Float16* __restrict__ y) {
  int idx = blockIdx.x * blockDim.x + threadIdx.x;  // B*HEADS*N
  int n = idx & (N_ - 1);
  int hh = (idx >> 10) & (HEADS_ - 1);
  int b = idx >> 13;

  const float alpha = alpha_p[0];
  const float om = 1.0f - alpha;
  const float w0 = alpha, w1 = alpha * om, w2 = w1 * om, w3 = w2 * om;

  float qsum0 = 0.f, qsum1 = 0.f, qsum2 = 0.f, qsum3 = 0.f;
  const size_t base0 =
      (size_t)b * (C_ * N_) + (size_t)(hh * DHEAD_) * N_ + (size_t)n;

#pragma unroll 4
  for (int d = 0; d < DHEAD_; ++d) {
    size_t base = base0 + (size_t)d * N_;
    float mem = 0.f, qs[T_];
#pragma unroll
    for (int t = 0; t < T_; ++t) {
      mem = mem * 0.5f + (float)qpre[(size_t)t * BCN_ + base];
      float s = (mem >= 1.0f) ? 1.0f : 0.0f;
      qs[t] = s;
      mem *= (1.0f - s);
    }
    qsum0 += qs[0] + w0 * qs[0];
    qsum1 += qs[1] + w1 * qs[0] + w0 * qs[1];
    qsum2 += qs[2] + w2 * qs[0] + w1 * qs[1] + w0 * qs[2];
    qsum3 += qs[3] + w3 * qs[0] + w2 * qs[1] + w1 * qs[2] + w0 * qs[3];
  }

  float at[T_];
  {
    float qq[T_] = {qsum0, qsum1, qsum2, qsum3};
    float mem = 0.f;
#pragma unroll
    for (int t = 0; t < T_; ++t) {
      mem = mem * 0.5f + qq[t];
      float s = (mem >= 0.5f) ? 1.0f : 0.0f;
      at[t] = s;
      mem *= (1.0f - s);
    }
  }

#pragma unroll 4
  for (int d = 0; d < DHEAD_; ++d) {
    size_t base = base0 + (size_t)d * N_;
    float mem = 0.f;
#pragma unroll
    for (int t = 0; t < T_; ++t) {
      mem = mem * 0.5f + (float)kpre[(size_t)t * BCN_ + base];
      float s = (mem >= 1.0f) ? 1.0f : 0.0f;
      mem *= (1.0f - s);
      y[(size_t)t * BCN_ + base] = (_Float16)(at[t] * s);
    }
  }
}

// -------------------------------------------------------------------------
extern "C" void kernel_launch(void* const* d_in, const int* in_sizes, int n_in,
                              void* d_out, int out_size, void* d_ws,
                              size_t ws_size, hipStream_t stream) {
  (void)in_sizes; (void)n_in; (void)out_size; (void)ws_size;
  const float* x      = (const float*)d_in[0];
  const float* q_w    = (const float*)d_in[1];
  const float* k_w    = (const float*)d_in[2];
  const float* proj_w = (const float*)d_in[3];
  const float* proj_b = (const float*)d_in[4];
  const float* qg = (const float*)d_in[5];
  const float* qb = (const float*)d_in[6];
  const float* qm = (const float*)d_in[7];
  const float* qv = (const float*)d_in[8];
  const float* kg = (const float*)d_in[9];
  const float* kb = (const float*)d_in[10];
  const float* km = (const float*)d_in[11];
  const float* kv = (const float*)d_in[12];
  const float* pg = (const float*)d_in[13];
  const float* pb = (const float*)d_in[14];
  const float* pm = (const float*)d_in[15];
  const float* pv = (const float*)d_in[16];
  const float* alpha = (const float*)d_in[17];

  // Workspace carve-up (256B aligned).
  char* ws = (char*)d_ws;
  size_t off = 0;
  auto carve = [&](size_t bytes) {
    void* p = ws + off;
    off = (off + bytes + 255) & ~(size_t)255;
    return p;
  };
  _Float16* wq16  = (_Float16*)carve((size_t)C_ * C_ * sizeof(_Float16));
  _Float16* wk16  = (_Float16*)carve((size_t)C_ * C_ * sizeof(_Float16));
  _Float16* wp16  = (_Float16*)carve((size_t)C_ * C_ * sizeof(_Float16));
  float*    params = (float*)carve(6 * C_ * sizeof(float));
  _Float16* xs    = (_Float16*)carve(TBCN_ * sizeof(_Float16));
  _Float16* qpre  = (_Float16*)carve(TBCN_ * sizeof(_Float16));
  _Float16* kpre  = (_Float16*)carve(TBCN_ * sizeof(_Float16));
  _Float16* ysp   = (_Float16*)carve(TBCN_ * sizeof(_Float16));

  snn_prep_channels_k<<<1, 256, 0, stream>>>(qg, qb, qm, qv, kg, kb, km, kv,
                                             pg, pb, pm, pv, proj_b, params);
  snn_convert_w_k<<<dim3(C_ * C_ / 256, 3), 256, 0, stream>>>(
      q_w, k_w, proj_w, wq16, wk16, wp16);
  snn_lif_input_k<<<BCN_ / 256, 256, 0, stream>>>(x, xs);

  dim3 g(N_ / 128, C_ / 128, T_ * B_);
  snn_gemm_bn_k<false><<<g, 256, 0, stream>>>(wq16, xs, params + 0 * C_,
                                              params + 1 * C_, (void*)qpre);
  snn_gemm_bn_k<false><<<g, 256, 0, stream>>>(wk16, xs, params + 2 * C_,
                                              params + 3 * C_, (void*)kpre);
  snn_attn_k<<<(B_ * HEADS_ * N_) / 256, 256, 0, stream>>>(qpre, kpre, alpha,
                                                           ysp);
  snn_gemm_bn_k<true><<<g, 256, 0, stream>>>(wp16, ysp, params + 4 * C_,
                                             params + 5 * C_, d_out);
}
